// DigitConvolutionalModel_11261404250328
// MI455X (gfx1250) — matrix operand order repaired and checked by reading, not compiled, not run on hardware
//
#include <hip/hip_runtime.h>

typedef __attribute__((ext_vector_type(16))) __bf16 v16bf;
typedef __attribute__((ext_vector_type(8)))  float  v8f;

#define KCHUNKS 25   // GEMM1 K = 800 = 25 * 32 (784 padded)
#define NTILES  7    // GEMM1 N = 112 = 7 * 16 (100 padded)

// RNE convert (prep kernels; cost irrelevant there)
__device__ __forceinline__ unsigned short f2bf(float f) {
    unsigned int u = __builtin_bit_cast(unsigned int, f);
    u += 0x7FFFu + ((u >> 16) & 1u);
    return (unsigned short)(u >> 16);
}
// fast converts for the hot path (round-half-up; < bf16 ULP difference)
__device__ __forceinline__ unsigned short f2bf_fast(float f) {
    unsigned int u = __builtin_bit_cast(unsigned int, f) + 0x8000u;
    return (unsigned short)(u >> 16);
}
__device__ __forceinline__ unsigned int pack2bf(float a, float b) {
    unsigned int ua = __builtin_bit_cast(unsigned int, a) + 0x8000u;
    unsigned int ub = __builtin_bit_cast(unsigned int, b) + 0x8000u;
    return __builtin_amdgcn_perm(ub, ua, 0x07060302u); // {ub[31:16], ua[31:16]}
}

// ---------------------------------------------------------------------------
// B-fragment element placement for v_wmma_f32_16x16x32_bf16 (K=32 chunk):
//   lane = ((kk/8)%2)*16 + n ; dword = (kk/16)*4 + (kk%8)/2 ; halfword = kk&1.
// ---------------------------------------------------------------------------

// Prep 1: fold conv into w1, emit bf16 B-fragments (K padded 784->800, N->112)
__global__ void prep_w1frag(const float* __restrict__ conv_w,
                            const float* __restrict__ w1,
                            unsigned short* __restrict__ w1frag) {
    int idx = blockIdx.x * blockDim.x + threadIdx.x;
    if (idx >= 800 * 112) return;
    int p = idx / 112;          // padded K index
    int j = idx % 112;          // padded N index
    float s = 0.f;
    if (p < 784 && j < 100) {
        int iy = p / 28, ix = p % 28;
        #pragma unroll
        for (int ky = 0; ky < 3; ++ky) {
            int r = iy - ky;
            if (r < 0 || r >= 26) continue;
            #pragma unroll
            for (int kx = 0; kx < 3; ++kx) {
                int c = ix - kx;
                if (c < 0 || c >= 26) continue;
                s += conv_w[ky * 3 + kx] * w1[(r * 26 + c) * 100 + j];
            }
        }
    }
    int kc = p >> 5, kk = p & 31;
    int nt = j >> 4, nn = j & 15;
    int lane = (((kk >> 3) & 1) << 4) | nn;
    size_t e = ((((size_t)kc * NTILES + nt) * 32 + lane) * 8 + (((kk >> 4) << 2) | ((kk & 7) >> 1))) * 2 + (kk & 1);
    w1frag[e] = f2bf(s);
}

// Prep 2: w2 (100x10 -> K=128,N=16 pad) and w3 (10x10 -> K=32,N=16 pad) fragments
__global__ void prep_small(const float* __restrict__ w2,
                           const float* __restrict__ w3,
                           unsigned short* __restrict__ w2frag,
                           unsigned short* __restrict__ w3frag) {
    int idx = blockIdx.x * blockDim.x + threadIdx.x;
    if (idx < 128 * 16) {
        int p = idx >> 4, j = idx & 15;
        float v = (p < 100 && j < 10) ? w2[p * 10 + j] : 0.f;
        int kc = p >> 5, kk = p & 31;
        int lane = (((kk >> 3) & 1) << 4) | j;
        size_t e = (((size_t)kc * 32 + lane) * 8 + (((kk >> 4) << 2) | ((kk & 7) >> 1))) * 2 + (kk & 1);
        w2frag[e] = f2bf(v);
    } else if (idx < 128 * 16 + 32 * 16) {
        int t = idx - 128 * 16;
        int p = t >> 4, j = t & 15;
        float v = (p < 10 && j < 10) ? w3[p * 10 + j] : 0.f;
        int lane = (((p >> 3) & 1) << 4) | j;
        size_t e = ((size_t)lane * 8 + (((p >> 4) << 2) | ((p & 7) >> 1))) * 2 + (p & 1);
        w3frag[e] = f2bf(v);
    }
}

// ---------------------------------------------------------------------------
// Fused forward pass, all three layers on WMMA.
// One WG = 256 threads = 8 waves = 128 batch rows.
// ---------------------------------------------------------------------------
__global__ __launch_bounds__(256)
void fused_mlp(const float* __restrict__ x,
               const unsigned short* __restrict__ w1frag,
               const unsigned short* __restrict__ w2frag,
               const unsigned short* __restrict__ w3frag,
               const float* __restrict__ b1,
               const float* __restrict__ b2,
               const float* __restrict__ b3,
               float* __restrict__ out) {
    // double-buffered A tile in fragment layout (xor bank swizzle): 2 x 8 KB
    __shared__ uint4 Au4[2][8 * 32 * 2];
    // h1: 8 waves x 16 rows x 128 K (pad) bf16, row stride 136 (conflict-free b128)
    __shared__ unsigned short Hbuf[8][16][136];
    // h2: 8 waves x 16 rows x 32 K (pad) bf16, row stride 40 (conflict-free b128)
    __shared__ unsigned short H2[8][16][40];

    const int tid  = threadIdx.x;
    const int wave = tid >> 5;
    const int lane = tid & 31;
    const int lm   = lane & 15;
    const int lh   = lane >> 4;
    const int swz  = (lane >> 3) & 1;
    const int rowBase = blockIdx.x * 128;

    // zero the K-padding of Hbuf (cols 112..135) and all of H2 (used cols set later)
    for (int t = tid; t < 8 * 16 * 24; t += 256) {
        int w = t / (16 * 24), rm = (t / 24) % 16, c = 112 + (t % 24);
        Hbuf[w][rm][c] = 0;
    }
    for (int t = tid; t < 8 * 16 * 40; t += 256) {
        int w = t / (16 * 40), rm = (t / 40) % 16, c = t % 40;
        H2[w][rm][c] = 0;
    }

    v8f acc[NTILES];
    #pragma unroll
    for (int nt = 0; nt < NTILES; ++nt)
        #pragma unroll
        for (int j = 0; j < 8; ++j) acc[nt][j] = 0.f;

    // per-thread staging geometry: rows {i*32 + tid/8}, 4 consecutive cols
    const int cb = (tid & 7) * 4;          // col base within 32-chunk
    const int rr = tid >> 3;               // row sub-index 0..31
    float stage[4][4];

    // NOTE: 784 % 4 == 0, so each float4 is either fully valid or fully padding.
    auto load_chunk = [&](int kc) {
        int gk = kc * 32 + cb;
        if (gk + 3 < 784) {
            #pragma unroll
            for (int i = 0; i < 4; ++i) {
                float4 v = *(const float4*)&x[(size_t)(rowBase + i * 32 + rr) * 784 + gk];
                stage[i][0] = v.x; stage[i][1] = v.y;
                stage[i][2] = v.z; stage[i][3] = v.w;
            }
        } else {
            #pragma unroll
            for (int i = 0; i < 4; ++i) {
                stage[i][0] = 0.f; stage[i][1] = 0.f;
                stage[i][2] = 0.f; stage[i][3] = 0.f;
            }
        }
    };
    auto store_chunk = [&](int buf) {
        const int h  = cb >> 4;                    // K half (0: K<16, 1: K>=16)
        const int dl = ((cb >> 3) & 1) << 4;       // lane-half from K
        const int d0 = (cb & 7) >> 1;              // dword 0 or 2 within group
        unsigned long long* As64 = (unsigned long long*)Au4[buf];
        #pragma unroll
        for (int i = 0; i < 4; ++i) {
            int r = i * 32 + rr;
            int mtile = r >> 4, lmr = r & 15;
            int s = (lmr >> 3) & 1;                // xor bank swizzle
            int phys = (mtile * 32 + (dl | lmr)) * 8 + ((h ^ s) << 2) + d0;  // even
            unsigned long long v =
                ((unsigned long long)pack2bf(stage[i][2], stage[i][3]) << 32) |
                pack2bf(stage[i][0], stage[i][1]);
            As64[phys >> 1] = v;
        }
    };

    // ---- GEMM1: software-pipelined, double-buffered ----
    load_chunk(0);
    store_chunk(0);
    __syncthreads();

    for (int kc = 0; kc < KCHUNKS; ++kc) {
        const int buf = kc & 1;
        if (kc + 1 < KCHUNKS) load_chunk(kc + 1);   // overlap HBM with WMMA

        union { uint4 u4[2]; v16bf v; } afrag;
        afrag.u4[0] = Au4[buf][(wave * 32 + lane) * 2 + swz];
        afrag.u4[1] = Au4[buf][(wave * 32 + lane) * 2 + (1 ^ swz)];

        #pragma unroll
        for (int nt = 0; nt < NTILES; ++nt) {
            union { uint4 u4[2]; v16bf v; } bfrag;
            const uint4* bp = (const uint4*)w1frag +
                              (((size_t)kc * NTILES + nt) * 32 + lane) * 2;
            bfrag.u4[0] = bp[0];
            bfrag.u4[1] = bp[1];
            acc[nt] = __builtin_amdgcn_wmma_f32_16x16x32_bf16(
                false, afrag.v, false, bfrag.v, (short)0, acc[nt], false, false);
        }
        if (kc + 1 < KCHUNKS) {
            store_chunk(buf ^ 1);
            __syncthreads();
        }
    }

    // ---- bias + ReLU -> h1 (bf16) in LDS.  C layout: (m = lh*8+j, n = nt*16+lm)
    #pragma unroll
    for (int nt = 0; nt < NTILES; ++nt) {
        int n = nt * 16 + lm;
        float bias = (n < 100) ? b1[n] : 0.f;
        #pragma unroll
        for (int j = 0; j < 8; ++j) {
            float v = acc[nt][j] + bias;
            v = v > 0.f ? v : 0.f;
            if (n >= 100) v = 0.f;
            Hbuf[wave][lh * 8 + j][n] = f2bf_fast(v);
        }
    }
    __syncthreads();

    // ---- layer 2 on WMMA: h1(16x128) @ w2(128x16), 4 K-chunks
    v8f acc2;
    #pragma unroll
    for (int j = 0; j < 8; ++j) acc2[j] = 0.f;
    #pragma unroll
    for (int kc2 = 0; kc2 < 4; ++kc2) {
        union { uint4 u4[2]; v16bf v; } afrag;
        afrag.u4[0] = *(const uint4*)&Hbuf[wave][lm][kc2 * 32 + lh * 8];
        afrag.u4[1] = *(const uint4*)&Hbuf[wave][lm][kc2 * 32 + 16 + lh * 8];
        union { uint4 u4[2]; v16bf v; } bfrag;
        const uint4* bp = (const uint4*)w2frag + ((size_t)kc2 * 32 + lane) * 2;
        bfrag.u4[0] = bp[0];
        bfrag.u4[1] = bp[1];
        acc2 = __builtin_amdgcn_wmma_f32_16x16x32_bf16(
            false, afrag.v, false, bfrag.v, (short)0, acc2, false, false);
    }
    {   // bias + ReLU -> h2 (bf16), n = lm, m = lh*8+j
        float bias2 = (lm < 10) ? b2[lm] : 0.f;
        #pragma unroll
        for (int j = 0; j < 8; ++j) {
            float v = acc2[j] + bias2;
            v = v > 0.f ? v : 0.f;
            if (lm >= 10) v = 0.f;
            H2[wave][lh * 8 + j][lm] = f2bf_fast(v);
        }
    }
    __syncthreads();

    // ---- layer 3 on WMMA: h2(16x32) @ w3(32x16), single chunk
    v8f acc3;
    #pragma unroll
    for (int j = 0; j < 8; ++j) acc3[j] = 0.f;
    {
        union { uint4 u4[2]; v16bf v; } afrag;
        afrag.u4[0] = *(const uint4*)&H2[wave][lm][lh * 8];
        afrag.u4[1] = *(const uint4*)&H2[wave][lm][16 + lh * 8];
        union { uint4 u4[2]; v16bf v; } bfrag;
        const uint4* bp = (const uint4*)w3frag + (size_t)lane * 2;
        bfrag.u4[0] = bp[0];
        bfrag.u4[1] = bp[1];
        acc3 = __builtin_amdgcn_wmma_f32_16x16x32_bf16(
            false, afrag.v, false, bfrag.v, (short)0, acc3, false, false);
    }
    if (lm < 10) {
        float bias3 = b3[lm];
        #pragma unroll
        for (int j = 0; j < 8; ++j) {
            size_t grow = (size_t)rowBase + wave * 16 + lh * 8 + j;
            out[grow * 10 + lm] = acc3[j] + bias3;
        }
    }
}

extern "C" void kernel_launch(void* const* d_in, const int* in_sizes, int n_in,
                              void* d_out, int out_size, void* d_ws, size_t ws_size,
                              hipStream_t stream) {
    const float* x      = (const float*)d_in[0];
    const float* conv_w = (const float*)d_in[1];
    const float* w1     = (const float*)d_in[2];
    const float* b1     = (const float*)d_in[3];
    const float* w2     = (const float*)d_in[4];
    const float* b2     = (const float*)d_in[5];
    const float* w3     = (const float*)d_in[6];
    const float* b3     = (const float*)d_in[7];
    float* out = (float*)d_out;

    // workspace layout (bf16): w1frag 89600, w2frag 2048, w3frag 512  (184320 B)
    unsigned short* w1frag = (unsigned short*)d_ws;
    unsigned short* w2frag = w1frag + (size_t)25 * 7 * 32 * 8 * 2;
    unsigned short* w3frag = w2frag + (size_t)4 * 32 * 8 * 2;

    prep_w1frag<<<(800 * 112 + 255) / 256, 256, 0, stream>>>(conv_w, w1, w1frag);
    prep_small<<<10, 256, 0, stream>>>(w2, w3, w2frag, w3frag);
    fused_mlp<<<65536 / 128, 256, 0, stream>>>(x, w1frag, w2frag, w3frag,
                                               b1, b2, b3, out);
}
